// ApproxAct_20925080666722
// MI455X (gfx1250) — matrix operand order repaired and checked by reading, not compiled, not run on hardware
//
#include <hip/hip_runtime.h>
#include <hip/hip_bf16.h>

// ---------------------------------------------------------------------------
// ApproxAct: out[i] = sum_j relu(x[i] - bias[j]) * weight[j]
//
// bias is sorted, so the sum is piecewise linear in x:
//   k = #{ j : bias[j] < x }
//   out = x * Sw[k] - Sbw[k],  Sw[k] = sum_{j<k} w[j],  Sbw[k] = sum_{j<k} w[j]*b[j]
// => O(log K) per element instead of O(K); the apply kernel is a pure
//    HBM stream (8 MB @ 23.3 TB/s ~ 0.34 us) with an 8-step LDS binary search.
// ---------------------------------------------------------------------------

#define NK    257   // NUM_KNOTS
#define NS    256   // number of slopes
#define NW    255   // number of weights / biases
#define SORTN 512   // padded bitonic sort size

typedef __attribute__((ext_vector_type(16))) _Float16 v16h;
typedef __attribute__((ext_vector_type(8)))  float    v8f;

// Workspace layout (floats):
//   [0   .. 254]  bias
//   [256 .. 511]  Sw   (Sw[0] = 0)
//   [512 .. 767]  Sbw  (Sbw[0] = 0)
//   [768 .. 1023] diag: WMMA intra-row scan of w      (16x16 f32)
//   [1024.. 1279] diag: WMMA intra-row scan of w*b    (16x16 f32)

__global__ void pwl_setup(const float* __restrict__ x_list,
                          const float* __restrict__ y_list,
                          float* __restrict__ tbl, int do_diag) {
  __shared__ float xs[SORTN];
  __shared__ float slope[NS];
  __shared__ float wgt[256];
  __shared__ float wb[256];

  const int tid = threadIdx.x;

  // Load, clip to [-100,100], pad with +inf sentinels.
  for (int i = tid; i < SORTN; i += 256) {
    float v = 1.0e30f;
    if (i < NK) {
      v = x_list[i];
      v = fminf(fmaxf(v, -100.0f), 100.0f);
    }
    xs[i] = v;
  }

  // Bitonic sort (ascending) of 512 elements with 256 threads.
  for (int k = 2; k <= SORTN; k <<= 1) {
    for (int j = k >> 1; j > 0; j >>= 1) {
      __syncthreads();
      for (int t = tid; t < SORTN; t += 256) {
        int ixj = t ^ j;
        if (ixj > t) {
          float a = xs[t], b = xs[ixj];
          bool asc = (t & k) == 0;
          bool swap = asc ? (a > b) : (a < b);
          if (swap) { xs[t] = b; xs[ixj] = a; }
        }
      }
    }
  }
  __syncthreads();

  if (tid == 0) {            // reference: x[0] = 2*LO, x[-1] = 2*HI (post-sort)
    xs[0]      = -200.0f;
    xs[NK - 1] =  200.0f;
  }
  __syncthreads();

  // slope[j] = (y[j+1]-y[j]) / (x[j+1]-x[j] + 1e-8), with y endpoint overrides:
  //   y[0]=y[1]=0, y[255]=x[255], y[256]=x[256]
  if (tid < NS) {
    int j = tid, j1 = tid + 1;
    float y0 = (j  < 2) ? 0.0f : ((j  >= NK - 2) ? xs[j]  : y_list[j]);
    float y1 = (j1 < 2) ? 0.0f : ((j1 >= NK - 2) ? xs[j1] : y_list[j1]);
    slope[j] = (y1 - y0) / (xs[j1] - xs[j] + 1e-8f);
  }
  __syncthreads();

  // weight[j] = slope[j+1]-slope[j]; bias[j] = xs[j+1]
  if (tid < NW) {
    float w = slope[tid + 1] - slope[tid];
    float b = xs[tid + 1];
    wgt[tid] = w;
    wb[tid]  = w * b;
    tbl[tid] = b;
  } else {
    wgt[tid] = 0.0f;
    wb[tid]  = 0.0f;
  }
  __syncthreads();

  // Authoritative fp32 exclusive prefix sums (255 adds -> negligible).
  if (tid == 0) {
    float aw = 0.0f, ab = 0.0f;
    tbl[256] = 0.0f;
    tbl[512] = 0.0f;
    for (int j = 0; j < NW; ++j) {
      aw += wgt[j];
      ab += wb[j];
      tbl[256 + j + 1] = aw;
      tbl[512 + j + 1] = ab;
    }
  }

  // ---- WMMA diagnostic scan (CDNA5 matrix pipe) -------------------------
  // A (16x32 f16): row m holds w[16m .. 16m+15], zero-padded in K=[16,32).
  // B (32x16 f16): upper-triangular ones U[K][c] = (K<16 && K<=c).
  // D = A x B gives all 16 intra-row inclusive scans in ONE v_wmma op.
  {
    const int lane = tid & 31;        // wave32 lane
    const int col  = lane & 15;
    const int hi   = lane >> 4;       // 0: lanes 0-15, 1: lanes 16-31
    const int m    = col;             // A-matrix row = lane % 16

    v16h a_w, a_wb, bmat;
#pragma unroll
    for (int h = 0; h < 16; ++h) {
      // A 16-bit layout: VGPR0-3 hold K = hi*8 + 0..7, VGPR4-7 hold K = 16 + hi*8 + 0..7
      int grp = h >> 3, within = h & 7;
      int K = grp * 16 + hi * 8 + within;
      float avw  = (K < 16) ? wgt[16 * m + K] : 0.0f;
      float avwb = (K < 16) ? wb[16 * m + K]  : 0.0f;
      a_w[h]  = (_Float16)avw;
      a_wb[h] = (_Float16)avwb;
      // B: same K order as A (VGPR0-3 K = hi*8 + 0..7, VGPR4-7 K = 16 + hi*8 + 0..7)
      int Krow = K;
      bmat[h] = (_Float16)((Krow < 16 && Krow <= col) ? 1.0f : 0.0f);
    }
    v8f c = {0.f, 0.f, 0.f, 0.f, 0.f, 0.f, 0.f, 0.f};
    v8f d_w  = __builtin_amdgcn_wmma_f32_16x16x32_f16(
        false, a_w,  false, bmat, (short)0, c, false, false);
    v8f d_wb = __builtin_amdgcn_wmma_f32_16x16x32_f16(
        false, a_wb, false, bmat, (short)0, c, false, false);

    if (do_diag && tid < 32) {        // wave-uniform branch; wave 0 stores
#pragma unroll
      for (int v = 0; v < 8; ++v) {
        int row = v + hi * 8;         // D layout: VGPR v = row v (+8 for hi lanes)
        tbl[768  + row * 16 + col] = d_w[v];
        tbl[1024 + row * 16 + col] = d_wb[v];
      }
    }
  }
}

// Branchless lower-bound: k = #{ j in [0,255) : sb[j] < xv }, k in [0,255].
__device__ __forceinline__ float pwl_eval(float xv,
                                          const float* __restrict__ sb,
                                          const float* __restrict__ sw,
                                          const float* __restrict__ sbw) {
  int k = 0;
#pragma unroll
  for (int s = 128; s > 0; s >>= 1) {
    int t = k + s;                     // t <= 255 always (steps sum to 255)
    k = (sb[t - 1] < xv) ? t : k;
  }
  return fmaf(xv, sw[k], -sbw[k]);
}

__global__ void pwl_apply(const float* __restrict__ x,
                          const float* __restrict__ tbl,
                          float* __restrict__ out, int n) {
  __shared__ float s_b[256];
  __shared__ float s_sw[256];
  __shared__ float s_sbw[256];

  const int tid = threadIdx.x;
  s_b[tid]   = (tid < NW) ? tbl[tid] : 1.0e30f;
  s_sw[tid]  = tbl[256 + tid];
  s_sbw[tid] = tbl[512 + tid];
  __syncthreads();

  const int nChunks = n >> 2;
  const int i = blockIdx.x * blockDim.x + tid;

  if (i < nChunks) {
    float4 xv = reinterpret_cast<const float4*>(x)[i];
    float4 ov;
    ov.x = pwl_eval(xv.x, s_b, s_sw, s_sbw);
    ov.y = pwl_eval(xv.y, s_b, s_sw, s_sbw);
    ov.z = pwl_eval(xv.z, s_b, s_sw, s_sbw);
    ov.w = pwl_eval(xv.w, s_b, s_sw, s_sbw);
    reinterpret_cast<float4*>(out)[i] = ov;
  } else if (i == nChunks) {
    for (int r = nChunks * 4; r < n; ++r)
      out[r] = pwl_eval(x[r], s_b, s_sw, s_sbw);
  }
}

extern "C" void kernel_launch(void* const* d_in, const int* in_sizes, int n_in,
                              void* d_out, int out_size, void* d_ws, size_t ws_size,
                              hipStream_t stream) {
  const float* x  = (const float*)d_in[0];
  const float* xl = (const float*)d_in[1];
  const float* yl = (const float*)d_in[2];
  float* out = (float*)d_out;
  float* tbl = (float*)d_ws;

  const int n = in_sizes[0];
  const int do_diag = (ws_size >= 1280 * sizeof(float)) ? 1 : 0;

  pwl_setup<<<1, 256, 0, stream>>>(xl, yl, tbl, do_diag);

  const int nChunks = n >> 2;
  const int threads = nChunks + 1;                // +1 thread slot for the tail
  const int blocks  = (threads + 255) / 256;
  pwl_apply<<<blocks, 256, 0, stream>>>(x, tbl, out, n);
}
